// VectorQuantize_74423193305312
// MI455X (gfx1250) — matrix operand order repaired and checked by reading, not compiled, not run on hardware
//
#include <hip/hip_runtime.h>

// ---------------------------------------------------------------------------
// VectorQuantize for MI455X (gfx1250, wave32, WMMA).
//
// Core: cosine-sim GEMM (16384 x 8192 x 512) + row argmax using
// v_wmma_f32_16x16x32_f16.  Compute-bound (~137 GFLOP vs ~150 MB traffic):
// f16 inputs + f32 accumulate; codebook (8 MB f16) and encodings (16 MB f16)
// are L2-resident (192 MB L2).  Inner loop is software-pipelined: a 4-deep
// rolling register buffer of B fragments keeps 8 global_load_b128 in flight
// under the WMMAs, and two accumulator chains halve the WMMA RAW depth.
// ---------------------------------------------------------------------------

typedef __attribute__((ext_vector_type(16))) _Float16 v16h;
typedef __attribute__((ext_vector_type(8)))  _Float16 v8h;
typedef __attribute__((ext_vector_type(4)))  _Float16 h4;
typedef __attribute__((ext_vector_type(8)))  float    v8f;

constexpr int Bsz = 8, Dd = 512, Tt = 2048, Kc = 8192;
constexpr int BT  = Bsz * Tt;                 // 16384 rows
constexpr size_t ZN = (size_t)Bsz * Dd * Tt;  // 8388608 elements per (B,D,T) tensor

union V16 { v16h v; v8h h[2]; };

// ---------------------------------------------------------------------------
// Kernel 1: L2-normalize codebook rows -> f16.  One wave per row (8 rows/block).
// ---------------------------------------------------------------------------
__global__ __launch_bounds__(256) void cb_norm_kernel(const float* __restrict__ cb,
                                                      _Float16* __restrict__ out) {
  const int wave = threadIdx.x >> 5, lane = threadIdx.x & 31;
  const int row  = blockIdx.x * 8 + wave;
  const float* src = cb + (size_t)row * Dd;
  float4 v[4];
  float ss = 0.f;
#pragma unroll
  for (int i = 0; i < 4; ++i) {
    v[i] = *(const float4*)(src + (i * 32 + lane) * 4);   // coalesced float4
    ss += v[i].x * v[i].x + v[i].y * v[i].y + v[i].z * v[i].z + v[i].w * v[i].w;
  }
#pragma unroll
  for (int m = 16; m >= 1; m >>= 1) ss += __shfl_xor(ss, m, 32);
  const float scale = 1.f / fmaxf(sqrtf(ss), 1e-12f);
  _Float16* dst = out + (size_t)row * Dd;
#pragma unroll
  for (int i = 0; i < 4; ++i) {
    h4 h = { (_Float16)(v[i].x * scale), (_Float16)(v[i].y * scale),
             (_Float16)(v[i].z * scale), (_Float16)(v[i].w * scale) };
    *(h4*)(dst + (i * 32 + lane) * 4) = h;
  }
}

// ---------------------------------------------------------------------------
// Kernel 2: per-(b,t) L2 norm of z over D, transpose to (BT, D) f16 encodings,
// fused with z_e = z output copy.  Block = 256 threads as (64 t, 4).
// ---------------------------------------------------------------------------
__global__ __launch_bounds__(256) void z_prep_kernel(const float* __restrict__ z,
                                                     _Float16* __restrict__ enc,
                                                     float* __restrict__ z_e_out) {
  __shared__ float tile[64][65];
  __shared__ float partial[4][64];
  __shared__ float scl[64];
  const int tx = threadIdx.x & 63, ty = threadIdx.x >> 6;
  const int b  = blockIdx.x >> 5;          // Tt/64 = 32 tiles per batch
  const int t0 = (blockIdx.x & 31) * 64;
  const float* zb = z + (size_t)b * Dd * Tt;

  // phase 1: sum of squares over D for 64 t's (coalesced along t)
  float ss = 0.f;
  for (int d = ty; d < Dd; d += 4) {
    const float v = zb[(size_t)d * Tt + t0 + tx];
    ss += v * v;
  }
  partial[ty][tx] = ss;
  __syncthreads();
  if (ty == 0) {
    float s = partial[0][tx] + partial[1][tx] + partial[2][tx] + partial[3][tx];
    scl[tx] = 1.f / fmaxf(sqrtf(s), 1e-12f);
  }
  __syncthreads();

  // phase 2: LDS transpose 64x64 tiles; write enc f16 + copy z_e
  for (int d0 = 0; d0 < Dd; d0 += 64) {
    for (int tt = ty; tt < 64; tt += 4) {
      float v = zb[(size_t)(d0 + tt) * Tt + t0 + tx];
      tile[tx][tt] = v;                                   // [t_local][d_local]
      z_e_out[(size_t)b * Dd * Tt + (size_t)(d0 + tt) * Tt + t0 + tx] = v;
    }
    __syncthreads();
    // enc row = b*Tt + t, col = d (contiguous f16 writes along d)
    for (int tt = ty; tt < 64; tt += 4)
      enc[(size_t)(b * Tt + t0 + tt) * Dd + d0 + tx] = (_Float16)(tile[tt][tx] * scl[tt]);
    __syncthreads();
  }
}

// ---------------------------------------------------------------------------
// Kernel 3: WMMA GEMM + argmax.  One workgroup (8 waves) per 16-row tile;
// each wave scans 1024 codes.  A-fragment layout (contiguous K, same order as B):
// lane l holds row M=l&15, kh=(l>>4)*16; v16h elems 0..7 -> K=kh..kh+7,
// elems 8..15 -> K=kh+8..kh+15.  B fragment: lane l holds column N=l&15,
// K=(l>>4)*16..+15 (32B contiguous).  C/D: vgpr r -> M=r+(l>>4)*8, lane->N.
// ---------------------------------------------------------------------------
__global__ __launch_bounds__(256) void vq_argmax_kernel(const _Float16* __restrict__ enc,
                                                        const _Float16* __restrict__ cb,
                                                        int* __restrict__ idx_out) {
  __shared__ float redV[8][16];
  __shared__ int   redI[8][16];
  const int tid  = threadIdx.x;
  const int wave = tid >> 5, lane = tid & 31;
  const int row0 = blockIdx.x * 16;
  const int kh16  = (lane >> 4) * 16;
  const int nlane = lane & 15;

  // A tile (16 rows x 512) held entirely in VGPRs (16 fragments x 8 VGPRs)
  v16h a[16];
  {
    const _Float16* arow = enc + (size_t)(row0 + nlane) * Dd + ((lane >> 4) * 16);
#pragma unroll
    for (int s = 0; s < 16; ++s) {
      V16 t;
      t.h[0] = *(const v8h*)(arow + s * 32);       // K = kh16..kh16+7
      t.h[1] = *(const v8h*)(arow + s * 32 + 8);   // K = kh16+8..kh16+15
      a[s] = t.v;
    }
  }

  float bv[8];
  int   bi[8];
#pragma unroll
  for (int r = 0; r < 8; ++r) { bv[r] = -1e30f; bi[r] = 0; }

  const int nBegin = wave * (Kc / 8), nEnd = nBegin + Kc / 8;

  // rolling 4-deep B-fragment pipeline (prefetches across tile boundaries;
  // the one-tile-past-the-end read of the last wave lands in the enc16
  // workspace region immediately after cb16 -- allocated, values unused).
  const _Float16* bptr = cb + (size_t)(nBegin + nlane) * Dd + kh16;
  v16h bb0 = *(const v16h*)(bptr);
  v16h bb1 = *(const v16h*)(bptr + 32);
  v16h bb2 = *(const v16h*)(bptr + 64);
  v16h bb3 = *(const v16h*)(bptr + 96);

  for (int n0 = nBegin; n0 < nEnd; n0 += 16) {
    v8f acc0 = {}, acc1 = {};
#pragma unroll
    for (int s = 0; s < 16; s += 4) {
      const int nofs = (s + 4 < 16) ? (s + 4) * 32 : 16 * Dd + ((s + 4) - 16) * 32;
      v16h p0 = *(const v16h*)(bptr + nofs);
      v16h p1 = *(const v16h*)(bptr + nofs + 32);
      v16h p2 = *(const v16h*)(bptr + nofs + 64);
      v16h p3 = *(const v16h*)(bptr + nofs + 96);
      acc0 = __builtin_amdgcn_wmma_f32_16x16x32_f16(false, a[s],     false, bb0,
                                                    (short)0, acc0, false, false);
      acc1 = __builtin_amdgcn_wmma_f32_16x16x32_f16(false, a[s + 1], false, bb1,
                                                    (short)0, acc1, false, false);
      acc0 = __builtin_amdgcn_wmma_f32_16x16x32_f16(false, a[s + 2], false, bb2,
                                                    (short)0, acc0, false, false);
      acc1 = __builtin_amdgcn_wmma_f32_16x16x32_f16(false, a[s + 3], false, bb3,
                                                    (short)0, acc1, false, false);
      bb0 = p0; bb1 = p1; bb2 = p2; bb3 = p3;
    }
    const int n = n0 + nlane;
#pragma unroll
    for (int r = 0; r < 8; ++r) {
      const float v = acc0[r] + acc1[r];
      if (v > bv[r]) { bv[r] = v; bi[r] = n; }   // strict > keeps lowest index
    }
    bptr += 16 * Dd;
  }

  // reduce across the 16 lanes holding the same rows (xor masks stay in-half)
#pragma unroll
  for (int r = 0; r < 8; ++r) {
    float v = bv[r]; int i = bi[r];
#pragma unroll
    for (int m = 8; m >= 1; m >>= 1) {
      float ov = __shfl_xor(v, m, 32);
      int   oi = __shfl_xor(i, m, 32);
      if (ov > v || (ov == v && oi < i)) { v = ov; i = oi; }
    }
    bv[r] = v; bi[r] = i;
  }
  if (nlane == 0) {
    const int half = lane >> 4;                            // rows half*8 + r
#pragma unroll
    for (int r = 0; r < 8; ++r) { redV[wave][half * 8 + r] = bv[r]; redI[wave][half * 8 + r] = bi[r]; }
  }
  __syncthreads();
  if (tid < 16) {
    float v = redV[0][tid]; int i = redI[0][tid];
#pragma unroll
    for (int w = 1; w < 8; ++w) {
      float ov = redV[w][tid]; int oi = redI[w][tid];
      if (ov > v || (ov == v && oi < i)) { v = ov; i = oi; }
    }
    idx_out[row0 + tid] = i;
  }
}

// ---------------------------------------------------------------------------
// Kernel 4: gather codebook[idx], transpose to (B,D,T), apply mask, write
// z_q and z_q_out (numerically identical), and indices as float.
// ---------------------------------------------------------------------------
__global__ __launch_bounds__(256) void gather_kernel(const float* __restrict__ cb,
                                                     const float* __restrict__ mask,
                                                     const int* __restrict__ idx,
                                                     float* __restrict__ zq,
                                                     float* __restrict__ zq_out,
                                                     float* __restrict__ idx_f) {
  __shared__ float tile[64][65];
  __shared__ int   idxs[64];
  __shared__ float maskv[64];
  const int tx = threadIdx.x & 63, ty = threadIdx.x >> 6;
  const int b  = blockIdx.x >> 5;
  const int t0 = (blockIdx.x & 31) * 64;
  if (ty == 0) {
    const int bt = b * Tt + t0 + tx;
    const int id = idx[bt];
    idxs[tx]  = id;
    maskv[tx] = mask[bt];                                  // mask is (B,1,T)
    idx_f[bt] = (float)id;
  }
  __syncthreads();
  for (int d0 = 0; d0 < Dd; d0 += 64) {
    for (int tt = ty; tt < 64; tt += 4)                    // coalesced along d
      tile[tt][tx] = cb[(size_t)idxs[tt] * Dd + d0 + tx];
    __syncthreads();
    for (int dd = ty; dd < 64; dd += 4) {                  // coalesced along t
      const float v = tile[tx][dd] * maskv[tx];
      const size_t off = (size_t)b * Dd * Tt + (size_t)(d0 + dd) * Tt + t0 + tx;
      zq[off] = v;
      zq_out[off] = v;
    }
    __syncthreads();
  }
}

// ---------------------------------------------------------------------------
extern "C" void kernel_launch(void* const* d_in, const int* in_sizes, int n_in,
                              void* d_out, int out_size, void* d_ws, size_t ws_size,
                              hipStream_t stream) {
  const float* z    = (const float*)d_in[0];   // (8, 512, 2048)
  const float* mask = (const float*)d_in[1];   // (8, 1, 2048)
  const float* cb   = (const float*)d_in[2];   // (8192, 512)

  float* out    = (float*)d_out;
  float* zq     = out;                         // (B,D,T)
  float* ze     = out + ZN;                    // (B,D,T)
  float* zq_out = out + 2 * ZN;                // (B,D,T)
  float* idx_f  = out + 3 * ZN;                // (B,T) indices as float

  // workspace: cb f16 (8 MB) | enc f16 (16 MB) | indices int (64 KB)
  char* ws = (char*)d_ws;
  _Float16* cb16  = (_Float16*)ws;
  _Float16* enc16 = (_Float16*)(ws + (size_t)Kc * Dd * 2);
  int*      idx_i = (int*)(ws + (size_t)Kc * Dd * 2 + (size_t)BT * Dd * 2);

  cb_norm_kernel<<<Kc / 8, 256, 0, stream>>>(cb, cb16);
  z_prep_kernel<<<BT / 64, 256, 0, stream>>>(z, enc16, ze);
  vq_argmax_kernel<<<BT / 16, 256, 0, stream>>>(enc16, cb16, idx_i);
  gather_kernel<<<BT / 64, 256, 0, stream>>>(cb, mask, idx_i, zq, zq_out, idx_f);
}
